// QuantGATModel_75788992905525
// MI455X (gfx1250) — matrix operand order repaired and checked by reading, not compiled, not run on hardware
//
#include <hip/hip_runtime.h>
#include <hip/hip_bf16.h>
#include <math.h>

// ---------------------------------------------------------------------------
// QuantGAT (2-layer GAT, H=2 heads, C=64) for MI455X / gfx1250 (wave32, WMMA)
// ---------------------------------------------------------------------------
#define Nn   100000
#define Ee   800000
#define INF_ 64
#define HIDc 64
#define HH   2
#define EDd  32
#define NEG_SLOPE 0.2f

typedef __attribute__((ext_vector_type(16))) _Float16 v16h;
typedef __attribute__((ext_vector_type(8)))  float    v8f;

__device__ __forceinline__ void atomAddF(float* p, float v) {
    __hip_atomic_fetch_add(p, v, __ATOMIC_RELAXED, __HIP_MEMORY_SCOPE_AGENT);
}
__device__ __forceinline__ void atomMaxF(float* p, float v) {
    __hip_atomic_fetch_max(p, v, __ATOMIC_RELAXED, __HIP_MEMORY_SCOPE_AGENT);
}

// ---------------- generic fill ----------------
__global__ void k_fill(float* __restrict__ p, float v, int n) {
    int i = blockIdx.x * blockDim.x + threadIdx.x;
    if (i < n) p[i] = v;
}

// ---------------- degree (segment count over src) ----------------
__global__ void k_degree(const int* __restrict__ src, float* __restrict__ deg, int e) {
    int i = blockIdx.x * blockDim.x + threadIdx.x;
    if (i < e) atomAddF(&deg[src[i]], 1.0f);
}

// ---------------- dscale[n] = (deg+1)^-a, both layers ----------------
__global__ void k_dscale(const float* __restrict__ deg,
                         const float* __restrict__ a1p, const float* __restrict__ a2p,
                         float* __restrict__ d1, float* __restrict__ d2, int n) {
    int i = blockIdx.x * blockDim.x + threadIdx.x;
    if (i >= n) return;
    float dv = deg[i] + 1.0f;
    d1[i] = powf(dv, -(*a1p));
    d2[i] = powf(dv, -(*a2p));
}

// ---------------- int8 per-tensor fake-quant of W -> f16 (single block) -----
__global__ __launch_bounds__(256)
void k_quant_w(const float* __restrict__ W, _Float16* __restrict__ Wq, int n) {
    __shared__ float red[256];
    float mx = 0.0f;
    for (int i = threadIdx.x; i < n; i += 256) mx = fmaxf(mx, fabsf(W[i]));
    red[threadIdx.x] = mx;
    __syncthreads();
    for (int s = 128; s > 0; s >>= 1) {
        if (threadIdx.x < s) red[threadIdx.x] = fmaxf(red[threadIdx.x], red[threadIdx.x + s]);
        __syncthreads();
    }
    float sc = fmaxf(red[0], 1e-8f) / 127.0f;
    for (int i = threadIdx.x; i < n; i += 256) {
        float q = rintf(W[i] / sc);                  // RNE, matches jnp.round
        q = fminf(fmaxf(q, -128.0f), 127.0f);
        Wq[i] = (_Float16)(q * sc);
    }
}

// ---------------- v[d*2+h] = sum_c We[(h*64+c)*32+d] * ae[h*64+c] -----------
__global__ void k_prep_v(const float* __restrict__ We, const float* __restrict__ ae,
                         float* __restrict__ v) {
    int i = threadIdx.x;                              // 64 threads: (d,h)
    if (i >= EDd * HH) return;
    int d = i >> 1, h = i & 1;
    float acc = 0.0f;
    for (int c = 0; c < HIDc; ++c) acc += We[(h * HIDc + c) * EDd + d] * ae[h * HIDc + c];
    v[i] = acc;
}

// ---------------- WMMA GEMM: H[N,128] = (X*dscale) @ Wq^T -------------------
// One wave -> one 16x16 tile. 8 waves/block cover the 128 output columns.
__global__ __launch_bounds__(256)
void k_gemm_wmma(const float* __restrict__ X, const float* __restrict__ dscale,
                 const _Float16* __restrict__ Wq, float* __restrict__ Hout) {
    const int wave    = threadIdx.x >> 5;             // 0..7 -> column tile
    const int lane    = threadIdx.x & 31;
    const int r       = lane & 15;
    const int hi      = lane >> 4;
    const int rowBase = blockIdx.x * 16;
    const int colBase = wave * 16;

    // B (32x16 f16): lane holds column colBase+r, K = 16*hi + i (+kk)
    const _Float16* wrow = Wq + (colBase + r) * 64 + hi * 16;
    v16h b0 = *(const v16h*)(wrow);                   // K in [16*hi, 16*hi+16)
    v16h b1 = *(const v16h*)(wrow + 32);              // K in [32+16*hi, ...)

    // A (16x32 f16): lane holds row rowBase+r; K = 16*hi + i (+kk)
    const int   row  = rowBase + r;
    const float s    = dscale[row];
    const float* xr  = X + row * 64;
    v16h a0, a1;
#pragma unroll
    for (int i = 0; i < 8; ++i) {
        a0[i]     = (_Float16)(xr[16 * hi + i]        * s);
        a0[8 + i] = (_Float16)(xr[16 * hi + 8 + i]    * s);
        a1[i]     = (_Float16)(xr[32 + 16 * hi + i]   * s);
        a1[8 + i] = (_Float16)(xr[40 + 16 * hi + i]   * s);
    }

    v8f c = {};
    c = __builtin_amdgcn_wmma_f32_16x16x32_f16(false, a0, false, b0, (short)0, c, false, false);
    c = __builtin_amdgcn_wmma_f32_16x16x32_f16(false, a1, false, b1, (short)0, c, false, false);

    // D: c[g] -> (M = g + 8*hi, N = r)
#pragma unroll
    for (int g = 0; g < 8; ++g)
        Hout[(rowBase + g + 8 * hi) * 128 + colBase + r] = c[g];
}

// ---------------- per-node attention coefficients ---------------------------
__global__ void k_attn_node(const float* __restrict__ H, const float* __restrict__ as,
                            const float* __restrict__ ad, float* __restrict__ asrc,
                            float* __restrict__ adst, int n2) {
    int i = blockIdx.x * blockDim.x + threadIdx.x;    // (node, head)
    if (i >= n2) return;
    int n = i >> 1, h = i & 1;
    const float* hp = H + n * 128 + h * 64;
    const float* p1 = as + h * 64;
    const float* p2 = ad + h * 64;
    float s1 = 0.0f, s2 = 0.0f;
#pragma unroll 8
    for (int c = 0; c < 64; ++c) { float v = hp[c]; s1 += v * p1[c]; s2 += v * p2[c]; }
    asrc[i] = s1;
    adst[i] = s2;
}

// ---------------- edge pass 1: alpha = leaky(asrc+adst+aedge); segmax -------
__global__ void k_edge_alpha(const int* __restrict__ src, const int* __restrict__ dst,
                             const float* __restrict__ EA, const float* __restrict__ v,
                             const float* __restrict__ asrc, const float* __restrict__ adst,
                             float* __restrict__ alphaE, float* __restrict__ m, int e) {
    int i = blockIdx.x * blockDim.x + threadIdx.x;
    if (i >= e) return;
    int s = src[i], d = dst[i];
    const float* ea = EA + i * EDd;
    float ae0 = 0.0f, ae1 = 0.0f;
#pragma unroll
    for (int k = 0; k < EDd; ++k) { float x = ea[k]; ae0 += x * v[k * 2]; ae1 += x * v[k * 2 + 1]; }
    float r0 = asrc[s * 2]     + adst[d * 2]     + ae0;
    float r1 = asrc[s * 2 + 1] + adst[d * 2 + 1] + ae1;
    r0 = (r0 >= 0.0f) ? r0 : NEG_SLOPE * r0;
    r1 = (r1 >= 0.0f) ? r1 : NEG_SLOPE * r1;
    alphaE[i * 2]     = r0;
    alphaE[i * 2 + 1] = r1;
    atomMaxF(&m[d * 2],     r0);
    atomMaxF(&m[d * 2 + 1], r1);
}

// ---------------- edge pass 2: e = exp(alpha - m[dst]); segsum denom --------
__global__ void k_edge_exp(const int* __restrict__ dst, const float* __restrict__ m,
                           float* __restrict__ alphaE, float* __restrict__ denom, int e2) {
    int i = blockIdx.x * blockDim.x + threadIdx.x;    // (edge, head)
    if (i >= e2) return;
    int e = i >> 1, h = i & 1;
    int d = dst[e];
    float mm = m[d * 2 + h];
    if (!isfinite(mm)) mm = 0.0f;
    float ex = expf(alphaE[i] - mm);
    alphaE[i] = ex;
    atomAddF(&denom[d * 2 + h], ex);
}

// ---------------- edge pass 3: agg[dst] += h[src] * alpha -------------------
__global__ void k_edge_msg(const int* __restrict__ src, const int* __restrict__ dst,
                           const float* __restrict__ H, const float* __restrict__ alphaE,
                           const float* __restrict__ denom, float* __restrict__ agg, int total) {
    int i = blockIdx.x * blockDim.x + threadIdx.x;    // (edge, channel 0..127)
    if (i >= total) return;
    int e = i >> 7, ch = i & 127, head = ch >> 6;
    int s = src[e], d = dst[e];
    float w = alphaE[e * 2 + head] / (denom[d * 2 + head] + 1e-16f);
    atomAddF(&agg[d * 128 + ch], H[s * 128 + ch] * w);
}

// ---------------- head-mean + bias; track per-tensor max|.| -----------------
__global__ __launch_bounds__(256)
void k_finalize(const float* __restrict__ agg, const float* __restrict__ bias,
                float* __restrict__ out, float* __restrict__ maxabs, int total) {
    int i = blockIdx.x * 256 + threadIdx.x;           // (node, channel 0..63)
    float t = 0.0f;
    if (i < total) {
        int n = i >> 6, c = i & 63;
        t = 0.5f * (agg[n * 128 + c] + agg[n * 128 + 64 + c]) + bias[c];
        out[i] = t;
    }
    __shared__ float red[256];
    red[threadIdx.x] = fabsf(t);
    __syncthreads();
    for (int s = 128; s > 0; s >>= 1) {
        if (threadIdx.x < s) red[threadIdx.x] = fmaxf(red[threadIdx.x], red[threadIdx.x + s]);
        __syncthreads();
    }
    if (threadIdx.x == 0) atomMaxF(maxabs, red[0]);
}

// ---------------- apply int8 fake-quant (+ optional ReLU) -------------------
__global__ void k_quant_apply(const float* __restrict__ tin, const float* __restrict__ maxabs,
                              float* __restrict__ tout, int n, int doRelu) {
    int i = blockIdx.x * blockDim.x + threadIdx.x;
    if (i >= n) return;
    float sc = fmaxf(*maxabs, 1e-8f) / 127.0f;
    float q  = rintf(tin[i] / sc);
    q = fminf(fmaxf(q, -128.0f), 127.0f);
    float val = q * sc;
    if (doRelu) val = fmaxf(val, 0.0f);
    tout[i] = val;
}

// ---------------------------------------------------------------------------
extern "C" void kernel_launch(void* const* d_in, const int* in_sizes, int n_in,
                              void* d_out, int out_size, void* d_ws, size_t ws_size,
                              hipStream_t stream) {
    const float* x   = (const float*)d_in[0];
    const int*   ei  = (const int*)  d_in[1];
    const float* ea  = (const float*)d_in[2];
    const float* a1p = (const float*)d_in[3];
    const float* W1  = (const float*)d_in[4];
    const float* We1 = (const float*)d_in[5];
    const float* as1 = (const float*)d_in[6];
    const float* ad1 = (const float*)d_in[7];
    const float* ae1 = (const float*)d_in[8];
    const float* b1  = (const float*)d_in[9];
    const float* a2p = (const float*)d_in[10];
    const float* W2  = (const float*)d_in[11];
    const float* We2 = (const float*)d_in[12];
    const float* as2 = (const float*)d_in[13];
    const float* ad2 = (const float*)d_in[14];
    const float* ae2 = (const float*)d_in[15];
    const float* b2  = (const float*)d_in[16];

    const int* srcI = ei;
    const int* dstI = ei + Ee;

    // ---- workspace carve-up (256B aligned) ----
    char*  base = (char*)d_ws;
    size_t cur  = 0;
    auto carve = [&](size_t bytes) -> char* {
        char* p = base + cur;
        cur += (bytes + 255) & ~size_t(255);
        return p;
    };
    float*    deg    = (float*)carve(sizeof(float) * Nn);
    float*    dsc1   = (float*)carve(sizeof(float) * Nn);
    float*    dsc2   = (float*)carve(sizeof(float) * Nn);
    float*    Hbuf   = (float*)carve(sizeof(float) * (size_t)Nn * 128);
    float*    H1buf  = (float*)carve(sizeof(float) * (size_t)Nn * 64);
    float*    agg    = (float*)carve(sizeof(float) * (size_t)Nn * 128);
    float*    asrcB  = (float*)carve(sizeof(float) * Nn * 2);
    float*    adstB  = (float*)carve(sizeof(float) * Nn * 2);
    float*    mB     = (float*)carve(sizeof(float) * Nn * 2);
    float*    denomB = (float*)carve(sizeof(float) * Nn * 2);
    float*    alphaE = (float*)carve(sizeof(float) * (size_t)Ee * 2);
    float*    vedge  = (float*)carve(sizeof(float) * EDd * HH);
    float*    maxabs = (float*)carve(sizeof(float) * 4);
    _Float16* Wq     = (_Float16*)carve(sizeof(_Float16) * 128 * 64);

    const int B = 256;
    auto G = [](int n, int b) { return (n + b - 1) / b; };

    // ---- degrees + dscales (shared by both layers) ----
    k_fill  <<<G(Nn, B), B, 0, stream>>>(deg, 0.0f, Nn);
    k_degree<<<G(Ee, B), B, 0, stream>>>(srcI, deg, Ee);
    k_dscale<<<G(Nn, B), B, 0, stream>>>(deg, a1p, a2p, dsc1, dsc2, Nn);

    // ================= layer 1 =================
    k_quant_w  <<<1, 256, 0, stream>>>(W1, Wq, 128 * 64);
    k_prep_v   <<<1, 64, 0, stream>>>(We1, ae1, vedge);
    k_gemm_wmma<<<Nn / 16, 256, 0, stream>>>(x, dsc1, Wq, Hbuf);
    k_attn_node<<<G(Nn * 2, B), B, 0, stream>>>(Hbuf, as1, ad1, asrcB, adstB, Nn * 2);

    k_fill<<<G(Nn * 2, B), B, 0, stream>>>(mB, -INFINITY, Nn * 2);
    k_fill<<<G(Nn * 2, B), B, 0, stream>>>(denomB, 0.0f, Nn * 2);
    k_fill<<<G(Nn * 128, B), B, 0, stream>>>(agg, 0.0f, Nn * 128);

    k_edge_alpha<<<G(Ee, B), B, 0, stream>>>(srcI, dstI, ea, vedge, asrcB, adstB, alphaE, mB, Ee);
    k_edge_exp  <<<G(Ee * 2, B), B, 0, stream>>>(dstI, mB, alphaE, denomB, Ee * 2);
    k_edge_msg  <<<G(Ee * 128, B), B, 0, stream>>>(srcI, dstI, Hbuf, alphaE, denomB, agg, Ee * 128);

    k_fill       <<<1, 32, 0, stream>>>(maxabs, 0.0f, 1);
    k_finalize   <<<G(Nn * 64, B), B, 0, stream>>>(agg, b1, H1buf, maxabs, Nn * 64);
    k_quant_apply<<<G(Nn * 64, B), B, 0, stream>>>(H1buf, maxabs, H1buf, Nn * 64, /*relu=*/1);

    // ================= layer 2 =================
    k_quant_w  <<<1, 256, 0, stream>>>(W2, Wq, 128 * 64);
    k_prep_v   <<<1, 64, 0, stream>>>(We2, ae2, vedge);
    k_gemm_wmma<<<Nn / 16, 256, 0, stream>>>(H1buf, dsc2, Wq, Hbuf);
    k_attn_node<<<G(Nn * 2, B), B, 0, stream>>>(Hbuf, as2, ad2, asrcB, adstB, Nn * 2);

    k_fill<<<G(Nn * 2, B), B, 0, stream>>>(mB, -INFINITY, Nn * 2);
    k_fill<<<G(Nn * 2, B), B, 0, stream>>>(denomB, 0.0f, Nn * 2);
    k_fill<<<G(Nn * 128, B), B, 0, stream>>>(agg, 0.0f, Nn * 128);

    k_edge_alpha<<<G(Ee, B), B, 0, stream>>>(srcI, dstI, ea, vedge, asrcB, adstB, alphaE, mB, Ee);
    k_edge_exp  <<<G(Ee * 2, B), B, 0, stream>>>(dstI, mB, alphaE, denomB, Ee * 2);
    k_edge_msg  <<<G(Ee * 128, B), B, 0, stream>>>(srcI, dstI, Hbuf, alphaE, denomB, agg, Ee * 128);

    k_fill       <<<1, 32, 0, stream>>>(maxabs, 0.0f, 1);
    k_finalize   <<<G(Nn * 64, B), B, 0, stream>>>(agg, b2, H1buf, maxabs, Nn * 64);
    k_quant_apply<<<G(Nn * 64, B), B, 0, stream>>>(H1buf, maxabs, (float*)d_out, Nn * 64, /*relu=*/0);
}
